// AttnPillarPool_10788957847930
// MI455X (gfx1250) — matrix operand order repaired and checked by reading, not compiled, not run on hardware
//
#include <hip/hip_runtime.h>
#include <hip/hip_bf16.h>

#define DIM     128
#define PILLARS 6
#define HEADS   8
#define VOX     4
#define TP      16      // pillars per tile (one WMMA M)
#define LN_EPS  1e-5f

typedef __attribute__((ext_vector_type(16))) __bf16          v16bf;
typedef __attribute__((ext_vector_type(16))) unsigned short  v16u16;
typedef __attribute__((ext_vector_type(8)))  float           v8f;

static __device__ __forceinline__ unsigned short f2bf(float f) {
    unsigned u = __float_as_uint(f);
    u += 0x7FFFu + ((u >> 16) & 1u);          // round-to-nearest-even
    return (unsigned short)(u >> 16);
}
static __device__ __forceinline__ unsigned pack2bf(float a, float b) {
    return (unsigned)f2bf(a) | ((unsigned)f2bf(b) << 16);
}

// A-fragment loader: row-major bf16 tile in LDS, stride DIM elements.
// Per ISA 7.12.2, lane (m = L&15, half = L>>4) holds K = kt*32+half*8 .. +7
// in V0..V3 and K = kt*32+16+half*8 .. +7 in V4..V7 -> two contiguous b128s.
static __device__ __forceinline__ v16bf load_a_frag(const unsigned short* tile,
                                                    int m, int kt, int half) {
    const uint4* p = (const uint4*)(tile + m * DIM + kt * 32 + half * 8);
    union { uint4 q[2]; v16u16 v; } u;
    u.q[0] = p[0];          // K block 0..7
    u.q[1] = p[2];          // K block 16..23 (+32 bytes)
    return __builtin_bit_cast(v16bf, u.v);
}

// Pre-swizzle wq,wk,wv,wo (128x128 each, row = out dim) into per-lane
// B-fragment order: frag(nt,kt) lane L holds col n=nt*16+(L&15),
// K = kt*32 + (L>>4)*8 + (e&7) + (e>>3)*16 (e=0..15), contiguous 32B per lane.
__global__ void prep_weights_kernel(const float* __restrict__ in_w,
                                    const float* __restrict__ out_w,
                                    unsigned short* __restrict__ ws) {
    int tid  = blockIdx.x * blockDim.x + threadIdx.x;   // 65536 threads
    int e    = tid & 15;
    int lane = (tid >> 4) & 31;
    int f    = (tid >> 9) & 31;                          // nt*4 + kt
    int w    = tid >> 14;                                // 0..3
    int nt = f >> 2, kt = f & 3;
    int n = nt * 16 + (lane & 15);
    int k = kt * 32 + ((lane >> 4) << 3) + (e & 7) + ((e >> 3) << 4);
    float v = (w < 3) ? in_w[(w * DIM + n) * DIM + k] : out_w[n * DIM + k];
    ws[((w * 32 + f) * 32 + lane) * 16 + e] = f2bf(v);
}

__global__ void __launch_bounds__(128, 1)
attn_pillar_kernel(const float* __restrict__ feat,
                   const float* __restrict__ pos_emb,
                   const float* __restrict__ in_b,
                   const float* __restrict__ out_b,
                   const float* __restrict__ gamma,
                   const float* __restrict__ beta,
                   const int*   __restrict__ z_idx,
                   const unsigned short* __restrict__ wf,
                   float* __restrict__ out) {
    __shared__ __align__(16) float          srcLDS[TP * DIM];            // 8 KB (f32; becomes h in place)
    __shared__ __align__(16) unsigned short srcBF[TP * DIM];             // 4 KB
    __shared__ __align__(16) unsigned short keyLDS[PILLARS * TP * DIM];  // 24 KB bf16 (kv + pos)
    __shared__ __align__(16) unsigned short valLDS[PILLARS * TP * DIM];  // 24 KB bf16 (kv)
    __shared__ __align__(16) float          qLDS[TP * DIM];              // 8 KB
    __shared__ __align__(16) unsigned short ctxBF[TP * DIM];             // 4 KB
    __shared__ int validLDS[TP];
    __shared__ int zLDS[TP * VOX];

    const int t    = threadIdx.x;
    const int lane = t & 31;
    const int wave = t >> 5;
    const int half = lane >> 4;
    const int l16  = lane & 15;
    const long p0  = (long)blockIdx.x * TP;

    // ---------- phase 0: stage src max, key/value bf16 tiles, valid ---------
    if (t < TP) validLDS[t] = 0;
    if (t < TP * VOX) zLDS[t] = z_idx[p0 * VOX + t];
    {
        unsigned* key32 = (unsigned*)keyLDS;
        unsigned* val32 = (unsigned*)valLDS;
        #pragma unroll
        for (int i = 0; i < (PILLARS * TP * DIM / 2) / 128; ++i) {
            int idx  = t + i * 128;          // u32 index
            int row  = idx >> 6;             // (s*16 + p)
            int c2   = (idx & 63) << 1;      // element col
            int s    = row >> 4;
            float pa = pos_emb[s * DIM + c2];
            float pb = pos_emb[s * DIM + c2 + 1];
            key32[idx] = pack2bf(pa, pb);    // invalid slots: key = 0 + pos
            val32[idx] = 0u;                 // invalid slots: value = 0
        }
    }
    #pragma unroll
    for (int i = 0; i < 4; ++i) {            // 512 float4 jobs: src = rowmax
        int jj  = t + i * 128;
        int row = jj >> 5;
        int c4  = (jj & 31) << 2;
        const float* base = feat + (p0 + row) * VOX * DIM + c4;
        float4 a = *(const float4*)(base);
        float4 b = *(const float4*)(base + DIM);
        float4 c = *(const float4*)(base + 2 * DIM);
        float4 d = *(const float4*)(base + 3 * DIM);
        float4 m;
        m.x = fmaxf(fmaxf(a.x, b.x), fmaxf(c.x, d.x));
        m.y = fmaxf(fmaxf(a.y, b.y), fmaxf(c.y, d.y));
        m.z = fmaxf(fmaxf(a.z, b.z), fmaxf(c.z, d.z));
        m.w = fmaxf(fmaxf(a.w, b.w), fmaxf(c.w, d.w));
        *(float4*)(srcLDS + row * DIM + c4) = m;
        unsigned* sb = (unsigned*)(srcBF + row * DIM + c4);
        sb[0] = pack2bf(m.x, m.y);
        sb[1] = pack2bf(m.z, m.w);
    }
    __syncthreads();
    for (int i = 0; i < 16; ++i) {           // 2048 float4 jobs: scatter
        int jj   = t + i * 128;
        int prow = jj >> 5;                  // point row 0..63
        int c4   = (jj & 31) << 2;
        int p = prow >> 2;
        int z = zLDS[prow];
        const float* base = feat + (p0 * VOX + prow) * DIM + c4;
        float4 v  = *(const float4*)base;
        float4 pe = *(const float4*)(pos_emb + z * DIM + c4);
        unsigned* vd = (unsigned*)(valLDS + (z * TP + p) * DIM + c4);
        vd[0] = pack2bf(v.x, v.y);
        vd[1] = pack2bf(v.z, v.w);
        unsigned* kd = (unsigned*)(keyLDS + (z * TP + p) * DIM + c4);
        kd[0] = pack2bf(v.x + pe.x, v.y + pe.y);
        kd[1] = pack2bf(v.z + pe.z, v.w + pe.w);
        if ((jj & 31) == 0) atomicOr(&validLDS[p], 1 << z);
    }
    __syncthreads();

    // ---------- phase 1: q = (src @ wq^T + bq) * scale -> qLDS --------------
    #pragma unroll
    for (int q8 = 0; q8 < 2; ++q8) {
        int nt = wave * 2 + q8;
        int nglob = nt * 16 + l16;
        float bq = in_b[nglob];
        v8f acc;
        #pragma unroll
        for (int r = 0; r < 8; ++r) acc[r] = bq;
        #pragma unroll
        for (int kt = 0; kt < 4; ++kt) {
            v16bf a = load_a_frag(srcBF, l16, kt, half);
            v16bf b = __builtin_bit_cast(v16bf,
                *(const v16u16*)(wf + ((0 * 32 + nt * 4 + kt) * 32 + lane) * 16));
            acc = __builtin_amdgcn_wmma_f32_16x16x32_bf16(
                      false, a, false, b, (short)0, acc, false, false);
        }
        #pragma unroll
        for (int r = 0; r < 8; ++r)
            qLDS[(r + 8 * half) * DIM + nglob] = acc[r] * 0.25f;  // 1/sqrt(16)
    }
    __syncthreads();

    // ---------- phase 2: per-head fused k/v WMMA + softmax + ctx ------------
    #pragma unroll 1
    for (int hh = 0; hh < 2; ++hh) {
        int h = wave + hh * 4;
        int nglob = h * 16 + l16;
        float bk = in_b[DIM + nglob];
        float bv = in_b[2 * DIM + nglob];
        float qreg[8]; int vm[8];
        #pragma unroll
        for (int r = 0; r < 8; ++r) {
            int p = r + 8 * half;
            qreg[r] = qLDS[p * DIM + nglob];
            vm[r]   = validLDS[p];
        }
        float S[PILLARS][8];
        #pragma unroll
        for (int s = 0; s < PILLARS; ++s) {
            v8f acc = {};
            #pragma unroll
            for (int kt = 0; kt < 4; ++kt) {
                v16bf a = load_a_frag(keyLDS + s * TP * DIM, l16, kt, half);
                v16bf b = __builtin_bit_cast(v16bf,
                    *(const v16u16*)(wf + ((1 * 32 + h * 4 + kt) * 32 + lane) * 16));
                acc = __builtin_amdgcn_wmma_f32_16x16x32_bf16(
                          false, a, false, b, (short)0, acc, false, false);
            }
            #pragma unroll
            for (int r = 0; r < 8; ++r) {    // q.k dot across the 16 N lanes
                float part = (acc[r] + bk) * qreg[r];
                part += __shfl_xor(part, 1, 16);
                part += __shfl_xor(part, 2, 16);
                part += __shfl_xor(part, 4, 16);
                part += __shfl_xor(part, 8, 16);
                S[s][r] = ((vm[r] >> s) & 1) ? part : -3.0e38f;
            }
        }
        #pragma unroll
        for (int r = 0; r < 8; ++r) {        // masked softmax over s
            float m = S[0][r];
            #pragma unroll
            for (int s = 1; s < PILLARS; ++s) m = fmaxf(m, S[s][r]);
            float sum = 0.f;
            #pragma unroll
            for (int s = 0; s < PILLARS; ++s) { float e = __expf(S[s][r] - m); S[s][r] = e; sum += e; }
            float inv = 1.0f / sum;
            #pragma unroll
            for (int s = 0; s < PILLARS; ++s) S[s][r] *= inv;
        }
        float ctx[8];
        #pragma unroll
        for (int r = 0; r < 8; ++r) ctx[r] = 0.f;
        #pragma unroll
        for (int s = 0; s < PILLARS; ++s) {  // v head-tile + attn weighting
            v8f acc = {};
            #pragma unroll
            for (int kt = 0; kt < 4; ++kt) {
                v16bf a = load_a_frag(valLDS + s * TP * DIM, l16, kt, half);
                v16bf b = __builtin_bit_cast(v16bf,
                    *(const v16u16*)(wf + ((2 * 32 + h * 4 + kt) * 32 + lane) * 16));
                acc = __builtin_amdgcn_wmma_f32_16x16x32_bf16(
                          false, a, false, b, (short)0, acc, false, false);
            }
            #pragma unroll
            for (int r = 0; r < 8; ++r) ctx[r] += S[s][r] * acc[r];
        }
        #pragma unroll
        for (int r = 0; r < 8; ++r)          // bv folded once (sum attn == 1)
            ctxBF[(r + 8 * half) * DIM + nglob] = f2bf(ctx[r] + bv);
    }
    __syncthreads();

    // ---------- phase 3: h = ctx @ wo^T + bo + src (in place into srcLDS) ---
    #pragma unroll
    for (int q8 = 0; q8 < 2; ++q8) {
        int nt = wave * 2 + q8;
        int nglob = nt * 16 + l16;
        float bo = out_b[nglob];
        v8f acc;
        #pragma unroll
        for (int r = 0; r < 8; ++r) acc[r] = bo;
        #pragma unroll
        for (int kt = 0; kt < 4; ++kt) {
            v16bf a = load_a_frag(ctxBF, l16, kt, half);
            v16bf b = __builtin_bit_cast(v16bf,
                *(const v16u16*)(wf + ((3 * 32 + nt * 4 + kt) * 32 + lane) * 16));
            acc = __builtin_amdgcn_wmma_f32_16x16x32_bf16(
                      false, a, false, b, (short)0, acc, false, false);
        }
        #pragma unroll
        for (int r = 0; r < 8; ++r) {
            int idx = (r + 8 * half) * DIM + nglob;
            srcLDS[idx] = acc[r] + srcLDS[idx];   // each element owned by one thread
        }
    }
    __syncthreads();

    // ---------- phase 4: LayerNorm + store ----------------------------------
    {
        int row = t >> 3;              // 16 rows, 8 threads each
        int seg = t & 7;               // 16 cols per thread
        const float* hrow = srcLDS + row * DIM + seg * 16;
        float vals[16];
        float sum = 0.f, sq = 0.f;
        #pragma unroll
        for (int i = 0; i < 16; ++i) { float v = hrow[i]; vals[i] = v; sum += v; sq += v * v; }
        sum += __shfl_xor(sum, 1, 8);  sq += __shfl_xor(sq, 1, 8);
        sum += __shfl_xor(sum, 2, 8);  sq += __shfl_xor(sq, 2, 8);
        sum += __shfl_xor(sum, 4, 8);  sq += __shfl_xor(sq, 4, 8);
        float mu  = sum * (1.0f / DIM);
        float var = sq * (1.0f / DIM) - mu * mu;
        float rs  = rsqrtf(var + LN_EPS);
        float* orow = out + (p0 + row) * DIM + seg * 16;
        #pragma unroll
        for (int i = 0; i < 16; ++i) {
            int c = seg * 16 + i;
            orow[i] = (vals[i] - mu) * rs * gamma[c] + beta[c];
        }
    }
}

extern "C" void kernel_launch(void* const* d_in, const int* in_sizes, int n_in,
                              void* d_out, int out_size, void* d_ws, size_t ws_size,
                              hipStream_t stream) {
    const float* feat  = (const float*)d_in[0];
    const float* pos   = (const float*)d_in[1];
    const float* in_w  = (const float*)d_in[2];
    const float* in_b  = (const float*)d_in[3];
    const float* out_w = (const float*)d_in[4];
    const float* out_b = (const float*)d_in[5];
    const float* gamma = (const float*)d_in[6];
    const float* beta  = (const float*)d_in[7];
    const int*   z_idx = (const int*)d_in[9];
    unsigned short* wf = (unsigned short*)d_ws;   // 128 KB of bf16 B-fragments

    int n_points = in_sizes[0] / DIM;
    int P = n_points / VOX;
    int tiles = P / TP;                            // P = 100000 -> 6250 exact

    prep_weights_kernel<<<256, 256, 0, stream>>>(in_w, out_w, wf);
    attn_pillar_kernel<<<tiles, 128, 0, stream>>>(feat, pos, in_b, out_b,
                                                  gamma, beta, z_idx, wf,
                                                  (float*)d_out);
}
